// GraphicalCVAE_20109036879950
// MI455X (gfx1250) — matrix operand order repaired and checked by reading, not compiled, not run on hardware
//
#include <hip/hip_runtime.h>
#include <hip/hip_bf16.h>
#include <cstring>
#include <cstdint>

// ---------------------------------------------------------------------------
// GraphicalCVAE forward on gfx1250 (CDNA5) using v_wmma_f32_16x16x32_f16 and
// the Tensor Data Mover (tensor_load_to_lds) for input staging.
//
//  * cvae_pack: packs ~544 tiny f32 parameter arrays into 176 ready-to-load
//    32x16 f16 WMMA B-fragments + bias vectors in d_ws. Encoder+prior of each
//    z-node fused block-diagonally (4 WMMA layers); mu/lv heads fused into
//    head-fragment columns.
//  * cvae_main: each wave owns 16 batch rows. Data tile (16x104 f32) staged
//    into LDS by one TDM descriptor per wave (TENSORcnt). Per MLP layer:
//    build f16 A fragment, load packed B fragment (32B/lane), one WMMA,
//    bias+ReLU epilogue into an f16 16x16 LDS activation tile (so hidden
//    A-builds are a single 16B LDS load per lane). Heads write a 16x120 f32
//    LDS output tile; reparameterization uses __expf; write-back coalesced.
//
// Layout assumptions (compile-only; cannot be runtime-verified):
//  * Input flattening = jax tree_leaves order (dict keys sorted at every
//    level): [data, eps, de(28 y sorted; w0,b0,w1,b1,w2,b2,out w,b),
//    en(16 z sorted; body..., lv w,b, mu w,b), pn(same as en)].
//  * 16-bit WMMA A layout per ISA table; B packed in the same table order:
//    lane l holds column n=l&15, elem s<8 at K=8*(l>>4)+s, s>=8 at K=16+8*(l>>4)+s-8;
//    VGPR v holds two consecutive K (elements 2v,2v+1).
//  * TDM D# bitfields per CDNA5 ISA; LDS offset =
//    low 32 bits of generic shared pointer (ISA §10.2 aperture mapping).
//  * This toolchain exposes the 6-arg tensor_load_to_lds builtin
//    (g0 v4u, g1 v8i, g2 v4i, g3 v4i, g4 v8i, cpol) — g2..g4 zero for <=2D.
// ---------------------------------------------------------------------------

typedef __attribute__((ext_vector_type(16))) _Float16 v16h;
typedef __attribute__((ext_vector_type(8)))  _Float16 v8h;
typedef __attribute__((ext_vector_type(8)))  float    v8f;
typedef __attribute__((ext_vector_type(4)))  unsigned int v4u;
typedef __attribute__((ext_vector_type(8)))  int      v8i;
typedef __attribute__((ext_vector_type(4)))  int      v4i;

struct Sub { int w, b, k0, n0, K, N; };   // sub-block of a packed fragment

#define NFRAG   176
#define NNODE   44          // 16 enpn + 28 decoder nodes
#define NPTRMAX 600
#define WAVES   8
#define WSTRIDE 3968        // per-wave LDS floats: data 1664 | act16 128 | zs 256 | out 1920

// d_ws layout (bytes)
#define OFF_W    0                         // 176*512 f16  = 180224
#define OFF_B    180224                    // 176*16  f32  = 11264
#define OFF_TAB  191488                    // 44*32   i8   = 1408
#define OFF_PTR  192896                    // 600*8        = 4800
#define OFF_DESC 197696                    // 176*4 Sub(24)= 16896

// ----------------------------- pack kernel ---------------------------------
__global__ void cvae_pack(const unsigned long long* __restrict__ ptrs,
                          const Sub* __restrict__ descs,
                          _Float16* __restrict__ W, float* __restrict__ B) {
  int f = blockIdx.x;
  for (int e = threadIdx.x; e < 512; e += blockDim.x) {
    int lane = e >> 4, t = e & 15;
    int k = ((t >> 3) << 4) + ((lane >> 4) << 3) + (t & 7);  // K index of this B element
    int n = lane & 15;                     // column
    float val = 0.f;
    #pragma unroll
    for (int s = 0; s < 4; ++s) {
      Sub d = descs[f * 4 + s];
      if (d.N > 0 && k >= d.k0 && k < d.k0 + d.K && n >= d.n0 && n < d.n0 + d.N)
        val = ((const float*)ptrs[d.w])[(k - d.k0) * d.N + (n - d.n0)];
    }
    W[(size_t)f * 512 + e] = (_Float16)val;
  }
  if (threadIdx.x < 16) {
    int n = threadIdx.x;
    float bv = 0.f;
    #pragma unroll
    for (int s = 0; s < 4; ++s) {
      Sub d = descs[f * 4 + s];
      if (d.N > 0 && n >= d.n0 && n < d.n0 + d.N)
        bv = ((const float*)ptrs[d.b])[n - d.n0];
    }
    B[f * 16 + n] = bv;
  }
}

// ----------------------------- main kernel ---------------------------------
__device__ __forceinline__ v8f do_wmma(v16h a, const _Float16* Wp, int f, int lane) {
  const v16h* bp = (const v16h*)(Wp + (size_t)f * 512);
  v16h b = bp[lane];                       // 32B per lane, L2-hot
  v8f c = {};
  return __builtin_amdgcn_wmma_f32_16x16x32_f16(false, a, false, b,
                                                (short)0, c, false, false);
}

// layer-0 A build: gather via node table (>=0: data col, -1: zero, <=-2: zsamp)
__device__ __forceinline__ v16h buildA_tab(const signed char* tab,
                                           const float* wdata, const float* wzs,
                                           int mr, int kh) {
  v16h a;
  #pragma unroll
  for (int v = 0; v < 8; ++v) {
    int kb = ((v >> 2) << 4) + (kh << 3) + ((v & 3) << 1);
    #pragma unroll
    for (int u = 0; u < 2; ++u) {
      int k = kb + u;
      int t = tab[k];
      float x;
      if (t >= 0)       x = wdata[mr * 104 + t];
      else if (t == -1) x = 0.f;
      else              x = wzs[mr * 16 + (-2 - t)];
      a[2 * v + u] = (_Float16)x;
    }
  }
  return a;
}

// hidden-layer A build: activation tile is f16 and fully written (padded cols
// are 0).  Lane's low-half A elements are exactly 8 contiguous f16 at
// act16[mr*16 + kh*8] -> one 16B ds load; high half (K>=16) is zero padding.
__device__ __forceinline__ v16h buildA_act16(const _Float16* act16, int mr, int kh) {
  v8h lo = *(const v8h*)(act16 + mr * 16 + kh * 8);
  v16h a;
  #pragma unroll
  for (int i = 0; i < 8; ++i) a[i] = lo[i];
  #pragma unroll
  for (int i = 8; i < 16; ++i) a[i] = (_Float16)0.f;
  return a;
}

__device__ __forceinline__ void epi_relu(v8f d, const float* Bp, int f,
                                         _Float16* act16, int nn, int kh) {
  float bb = Bp[f * 16 + nn];
  #pragma unroll
  for (int r = 0; r < 8; ++r) {
    float x = fmaxf(d[r] + bb, 0.f);
    act16[(r + 8 * kh) * 16 + nn] = (_Float16)x;
  }
  asm volatile("s_wait_dscnt 0x0" ::: "memory");
}

__global__ void __launch_bounds__(256)
cvae_main(const float* __restrict__ data, const float* __restrict__ eps,
          float* __restrict__ out, const _Float16* __restrict__ Wp,
          const float* __restrict__ Bp, const signed char* __restrict__ gtab,
          int rows) {
  __shared__ float sm[WAVES * WSTRIDE];
  __shared__ signed char stab[NNODE * 32];
  for (int t = threadIdx.x; t < NNODE * 32; t += blockDim.x) stab[t] = gtab[t];
  __syncthreads();

  int lane = threadIdx.x & 31, wid = threadIdx.x >> 5;
  int tile = blockIdx.x * WAVES + wid;
  int rowbase = tile * 16;
  if (rowbase >= rows) return;

  float*     wdata = sm + wid * WSTRIDE;
  _Float16*  act16 = (_Float16*)(wdata + 1664);
  float*     wzs   = wdata + 1664 + 128;
  float*     wout  = wzs + 256;

  // ---- stage 16x104 f32 data tile via the Tensor Data Mover --------------
  {
    unsigned lds_off = (unsigned)(size_t)(void*)wdata;        // LDS byte offset
    unsigned long long ga =
        (unsigned long long)(size_t)(data + (size_t)rowbase * 104);
    v4u g0;
    g0[0] = 1u;                                               // count=1, user D#
    g0[1] = lds_off;                                          // lds_addr
    g0[2] = (unsigned)(ga & 0xFFFFFFFFu);                     // global_addr lo
    g0[3] = (unsigned)((ga >> 32) & 0x1FFFFFFu) | (2u << 30); // addr hi | type=2
    v8i g1;
    g1[0] = (int)(2u << 16);                 // data_size = 2 (4 bytes)
    g1[1] = (int)((1664u & 0xFFFFu) << 16);  // tensor_dim0 lo16 (bits 63:48)
    g1[2] = (int)((1664u >> 16) | (1u << 16)); // dim0 hi | tensor_dim1 = 1
    g1[3] = (int)(1664u << 16);              // tile_dim0 = 1664 (bits 127:112)
    g1[4] = 0;                               // tile_dim1 = 0 (1D), tile_dim2 = 0
    g1[5] = 1664;                            // tensor_dim0_stride lo32
    g1[6] = 0;
    g1[7] = 0;
    v4i g2 = {0, 0, 0, 0};
    v4i g3 = {0, 0, 0, 0};
    v8i g4 = {0, 0, 0, 0, 0, 0, 0, 0};
    __builtin_amdgcn_tensor_load_to_lds(g0, g1, g2, g3, g4, 0);
    __builtin_amdgcn_s_wait_tensorcnt(0);
    asm volatile("" ::: "memory");
  }

  int mr = lane & 15;         // A-fragment row / D column index
  int kh = lane >> 4;         // half-lane selector
  int nn = lane & 15;         // D/N column index

  // ---- fused encoder+prior per z-node (z_all order: s0..s7, s0_net..s7_net)
  for (int j = 0; j < 16; ++j) {
    const signed char* tab = stab + j * 32;
    int f = 4 * j;
    v16h a = buildA_tab(tab, wdata, wzs, mr, kh);
    v8f  d = do_wmma(a, Wp, f + 0, lane); epi_relu(d, Bp, f + 0, act16, nn, kh);
    a = buildA_act16(act16, mr, kh);
    d = do_wmma(a, Wp, f + 1, lane);      epi_relu(d, Bp, f + 1, act16, nn, kh);
    a = buildA_act16(act16, mr, kh);
    d = do_wmma(a, Wp, f + 2, lane);      epi_relu(d, Bp, f + 2, act16, nn, kh);
    a = buildA_act16(act16, mr, kh);
    d = do_wmma(a, Wp, f + 3, lane);      // heads: n=0 mu, 1 lv, 2 pmu, 3 plv
    float bb = Bp[(f + 3) * 16 + nn];
    if (nn < 4) {
      int col = 56 + 16 * nn + j;          // 56+j / 72+j / 88+j / 104+j
      #pragma unroll
      for (int r = 0; r < 8; ++r)
        wout[(r + 8 * kh) * 120 + col] = d[r] + bb;
    }
    asm volatile("s_wait_dscnt 0x0" ::: "memory");
    if (lane < 16) {                       // reparameterization
      float mu = wout[lane * 120 + 56 + j];
      float lv = wout[lane * 120 + 72 + j];
      float ev = eps[(size_t)(rowbase + lane) * 16 + j];
      wzs[lane * 16 + j] = mu + __expf(0.5f * lv) * ev;
    }
    asm volatile("s_wait_dscnt 0x0" ::: "memory");
  }

  // ---- decoders (y_all order)
  for (int q = 0; q < 28; ++q) {
    const signed char* tab = stab + (16 + q) * 32;
    int f = 64 + 4 * q;
    v16h a = buildA_tab(tab, wdata, wzs, mr, kh);
    v8f  d = do_wmma(a, Wp, f + 0, lane); epi_relu(d, Bp, f + 0, act16, nn, kh);
    a = buildA_act16(act16, mr, kh);
    d = do_wmma(a, Wp, f + 1, lane);      epi_relu(d, Bp, f + 1, act16, nn, kh);
    a = buildA_act16(act16, mr, kh);
    d = do_wmma(a, Wp, f + 2, lane);      epi_relu(d, Bp, f + 2, act16, nn, kh);
    a = buildA_act16(act16, mr, kh);
    d = do_wmma(a, Wp, f + 3, lane);      // out head: cols 2q, 2q+1
    float bb = Bp[(f + 3) * 16 + nn];
    if (nn < 2) {
      #pragma unroll
      for (int r = 0; r < 8; ++r)
        wout[(r + 8 * kh) * 120 + 2 * q + nn] = d[r] + bb;
    }
    asm volatile("s_wait_dscnt 0x0" ::: "memory");
  }

  // ---- coalesced 16x120 write-back
  for (int t = lane; t < 16 * 120; t += 32)
    out[(size_t)rowbase * 120 + t] = wout[t];
}

// ------------------------- host-side metadata ------------------------------
static unsigned long long h_ptrs[NPTRMAX];
static Sub                h_desc[NFRAG * 4];
static signed char        h_tab[NNODE * 32];

static void build_meta() {
  memset(h_desc, 0, sizeof(h_desc));
  memset(h_tab, -1, sizeof(h_tab));
  auto SUB = [&](int f, int s, int w, int b, int k0, int n0, int K, int N) {
    h_desc[f * 4 + s] = Sub{w, b, k0, n0, K, N};
  };
  // --- fused encoder(EN 10,10,5)+prior(PN 5,5,5) nodes, z_all order
  for (int j = 0; j < 16; ++j) {
    int i = j & 7;  bool net = (j >= 8);
    int pos = net ? (2 * i + 1) : (2 * i);       // sorted key position
    int enb = 226 + 10 * pos, pnb = 386 + 10 * pos;
    int enl[20], ne = 0, pnl[4], np = 0;
    if (!net) {
      if (i == 0)       { int v[] = {0,1,2,3,28,29,42,43};            for (int x : v) enl[ne++] = x; }
      else if (i == 7)  { int v[] = {26,27};                          for (int x : v) enl[ne++] = x; }
      else { int v[] = {4*i-2,4*i-1, 4*i,4*i+1, 4*i,4*i+1, 4*i+2,4*i+3,
                        28+2*i,29+2*i, 42+2*i,43+2*i};                for (int x : v) enl[ne++] = x; }
      for (int x = 0; x < 4; ++x) enl[ne++] = 56 + 4 * i + x;
      for (int x = 0; x < 4; ++x) pnl[np++] = 56 + 4 * i + x;
    } else {
      if (i == 0)       { enl[ne++] = 28; enl[ne++] = 29; }
      else if (i == 7)  { enl[ne++] = 54; enl[ne++] = 55; }
      else { enl[ne++] = 28+2*i; enl[ne++] = 29+2*i; enl[ne++] = 40+2*i; enl[ne++] = 41+2*i; }
      enl[ne++] = 88 + 2 * i; enl[ne++] = 89 + 2 * i;
      pnl[np++] = 88 + 2 * i; pnl[np++] = 89 + 2 * i;
    }
    for (int t = 0; t < ne; ++t) h_tab[j * 32 + t]      = (signed char)enl[t];
    for (int t = 0; t < np; ++t) h_tab[j * 32 + ne + t] = (signed char)pnl[t];
    int f = 4 * j;
    SUB(f+0,0, enb+0,enb+1, 0,0,  ne,10);  SUB(f+0,1, pnb+0,pnb+1, ne,10, np,5);
    SUB(f+1,0, enb+2,enb+3, 0,0,  10,10);  SUB(f+1,1, pnb+2,pnb+3, 10,10, 5,5);
    SUB(f+2,0, enb+4,enb+5, 0,0,  10,5);   SUB(f+2,1, pnb+4,pnb+5, 10,5,  5,5);
    // heads: en mu(col0), en lv(col1), pn mu(col2), pn lv(col3); sorted
    // pytree order stores lv before mu -> lv at base+6/7, mu at base+8/9
    SUB(f+3,0, enb+8,enb+9, 0,0, 5,1);
    SUB(f+3,1, enb+6,enb+7, 0,1, 5,1);
    SUB(f+3,2, pnb+8,pnb+9, 5,2, 5,1);
    SUB(f+3,3, pnb+6,pnb+7, 5,3, 5,1);
  }
  // --- decoder nodes (DE 5,10,10 + out 2), y_all order
  for (int q = 0; q < 28; ++q) {
    int lst[16]; int c = 0; int pos;
    if (q < 14) {
      int i = q >> 1;
      if ((q & 1) == 0) {                      // c{i}_client
        pos = 4 * i;
        for (int x = 0; x < 4; ++x) lst[c++] = 56 + 4 * i + x;        // de_x: s_i
        int dy[] = {4*i+2,4*i+3, 28+2*i,29+2*i, 42+2*i,43+2*i};
        for (int x : dy) lst[c++] = x;                                 // de_y
        lst[c++] = -2 - i;                                             // z: s_i
      } else {                                 // c{i}_server
        pos = 4 * i + 3;
        for (int x = 0; x < 4; ++x) lst[c++] = 56 + 4 * (i + 1) + x;  // de_x: s_{i+1}
        if (i < 6) { lst[c++] = 4*i+4; lst[c++] = 4*i+5; }            // de_y
        lst[c++] = -2 - (i + 1);                                       // z: s_{i+1}
      }
    } else if (q < 21) {                       // c{i}_net_req
      int i = q - 14; pos = 4 * i + 1;
      lst[c++] = 88 + 2 * i; lst[c++] = 89 + 2 * i;
      lst[c++] = -2 - (8 + i);                                         // z: s_i_net
    } else {                                   // c{i}_net_resp
      int i = q - 21; pos = 4 * i + 2;
      lst[c++] = 90 + 2 * i; lst[c++] = 91 + 2 * i;                    // x_net s_{i+1}
      lst[c++] = -2 - (9 + i);                                         // z: s_{i+1}_net
    }
    int node = 16 + q;
    for (int t = 0; t < c; ++t) h_tab[node * 32 + t] = (signed char)lst[t];
    int base = 2 + 8 * pos;
    int f = 64 + 4 * q;
    SUB(f+0,0, base+0,base+1, 0,0, c,5);
    SUB(f+1,0, base+2,base+3, 0,0, 5,10);
    SUB(f+2,0, base+4,base+5, 0,0, 10,10);
    SUB(f+3,0, base+6,base+7, 0,0, 10,2);
  }
}

extern "C" void kernel_launch(void* const* d_in, const int* in_sizes, int n_in,
                              void* d_out, int out_size, void* d_ws, size_t ws_size,
                              hipStream_t stream) {
  (void)out_size; (void)ws_size;
  build_meta();
  int np = (n_in < NPTRMAX) ? n_in : NPTRMAX;
  for (int t = 0; t < np; ++t) h_ptrs[t] = (unsigned long long)d_in[t];

  char* ws = (char*)d_ws;
  _Float16*           Wp   = (_Float16*)(ws + OFF_W);
  float*              Bpk  = (float*)(ws + OFF_B);
  signed char*        Tab  = (signed char*)(ws + OFF_TAB);
  unsigned long long* Ptr  = (unsigned long long*)(ws + OFF_PTR);
  Sub*                Desc = (Sub*)(ws + OFF_DESC);

  (void)hipMemcpyAsync(Tab,  h_tab,  sizeof(h_tab),  hipMemcpyHostToDevice, stream);
  (void)hipMemcpyAsync(Ptr,  h_ptrs, (size_t)np * 8, hipMemcpyHostToDevice, stream);
  (void)hipMemcpyAsync(Desc, h_desc, sizeof(h_desc), hipMemcpyHostToDevice, stream);

  cvae_pack<<<NFRAG, 256, 0, stream>>>(Ptr, Desc, Wp, Bpk);

  int rows   = in_sizes[0] / 104;
  int tiles  = rows / 16;
  int blocks = (tiles + WAVES - 1) / WAVES;
  cvae_main<<<blocks, 256, 0, stream>>>((const float*)d_in[0], (const float*)d_in[1],
                                        (float*)d_out, Wp, Bpk, Tab, rows);
}
